// Block_60490319396871
// MI455X (gfx1250) — matrix operand order repaired and checked by reading, not compiled, not run on hardware
//
#include <hip/hip_runtime.h>
#include <hip/hip_bf16.h>
#include <stdint.h>

// Problem dimensions (fixed by the reference)
#define Bx   2
#define Lx   2048
#define DM   1024
#define DI   2048
#define DST  16
#define DCV  4
#define DTR  64
#define Rn   (Bx * Lx)            // 4096 rows
#define NXD  (DTR + 2 * DST)      // 96

typedef __bf16 bf16_t;
typedef __attribute__((ext_vector_type(16))) __bf16        v16bf;
typedef __attribute__((ext_vector_type(8)))  float         v8f;
typedef __attribute__((ext_vector_type(8)))  unsigned int  v8u;
typedef __attribute__((ext_vector_type(4)))  unsigned int  u32x4;
typedef __attribute__((ext_vector_type(8)))  int           i32x8;
typedef __attribute__((ext_vector_type(4)))  int           i32x4;

#if defined(__AMDGCN__) && __has_builtin(__builtin_amdgcn_tensor_load_to_lds) && \
    __has_builtin(__builtin_amdgcn_s_wait_tensorcnt)
#define HAVE_TDM 1
#else
#define HAVE_TDM 0
#endif

// ---------------- workspace layout (bytes) ----------------
constexpr size_t OFF_XN    = 0;                                   // bf16 Rn*DM
constexpr size_t OFF_WTIN  = OFF_XN    + (size_t)Rn * DM * 2;     // bf16 (2*DI)*DM
constexpr size_t OFF_WTX   = OFF_WTIN  + (size_t)2 * DI * DM * 2; // bf16 NXD*DI
constexpr size_t OFF_WTDT  = OFF_WTX   + (size_t)NXD * DI * 2;    // bf16 DI*DTR
constexpr size_t OFF_WTO   = OFF_WTDT  + (size_t)DI * DTR * 2;    // bf16 DM*DI
constexpr size_t OFF_XZ    = OFF_WTO   + (size_t)DM * DI * 2;     // f32  Rn*2*DI
constexpr size_t OFF_XC    = OFF_XZ    + (size_t)Rn * 2 * DI * 4; // f32  Rn*DI
constexpr size_t OFF_XCB   = OFF_XC    + (size_t)Rn * DI * 4;     // bf16 Rn*DI
constexpr size_t OFF_XDBL  = OFF_XCB   + (size_t)Rn * DI * 2;     // f32  Rn*NXD
constexpr size_t OFF_DTB   = OFF_XDBL  + (size_t)Rn * NXD * 4;    // bf16 Rn*DTR
constexpr size_t OFF_DELTA = OFF_DTB   + (size_t)Rn * DTR * 2;    // f32  Rn*DI
constexpr size_t OFF_YRAW  = OFF_DELTA + (size_t)Rn * DI * 4;     // f32  Rn*DI
constexpr size_t OFF_YB    = OFF_YRAW  + (size_t)Rn * DI * 4;     // bf16 Rn*DI

// =====================================================================
// WMMA fragment gathers (ISA 7.12.2 16-bit layouts). Pairs are
// contiguous -> dword (and mergeable b128) loads from global or LDS.
// =====================================================================
__device__ __forceinline__ v16bf frag_a_from(const void* rowbase_bytes, int row_stride_bytes) {
  const int lane = threadIdx.x & 31;
  const int half = lane >> 4;
  const char* base = (const char*)rowbase_bytes + (size_t)(lane & 15) * row_stride_bytes;
  v8u u;
#pragma unroll
  for (int v = 0; v < 8; ++v) {
    const int kb = ((v & 4) << 2) + half * 8 + (v & 3) * 2;  // A: K pair index
    u[v] = *(const unsigned int*)(base + kb * 2);
  }
  union { v8u u; v16bf b; } cv; cv.u = u;
  return cv.b;
}
__device__ __forceinline__ v16bf frag_b_from(const void* rowbase_bytes, int row_stride_bytes) {
  const int lane = threadIdx.x & 31;
  const int half = lane >> 4;
  const char* base = (const char*)rowbase_bytes + (size_t)(lane & 15) * row_stride_bytes;
  v8u u;
#pragma unroll
  for (int v = 0; v < 8; ++v) {
    const int kb = ((v & 4) << 2) + half * 8 + (v & 3) * 2;  // B: K pair index
    u[v] = *(const unsigned int*)(base + kb * 2);
  }
  union { v8u u; v16bf b; } cv; cv.u = u;
  return cv.b;
}

// =====================================================================
// TDM descriptor issue: 2D tile (tile_rows x 32 bf16) from a row-major
// matrix (row stride = ldk elements) into LDS at lds_addr. D# packing
// per ISA 8.3/8.4: data_size=1 (2B), type=2, count=1.
// =====================================================================
__device__ __forceinline__ void tdm_load_tile(unsigned int lds_addr, const bf16_t* gsrc,
                                              unsigned int tensor_d0, unsigned int tensor_d1,
                                              unsigned int tile_rows, unsigned int ldk) {
#if HAVE_TDM
  const unsigned long long ga = (unsigned long long)(uintptr_t)gsrc;
  u32x4 g0;
  g0[0] = 1u;                                                // count=1 (valid), user mode
  g0[1] = lds_addr;                                          // lds_addr
  g0[2] = (unsigned int)ga;                                  // global_addr[31:0]
  g0[3] = (unsigned int)((ga >> 32) & 0x1FFFFFFull) | (2u << 30);  // addr[56:32] | type=2
  i32x8 g1;
  g1[0] = (int)(1u << 16);                                   // data_size=1 -> 2 bytes
  g1[1] = (int)((tensor_d0 & 0xFFFFu) << 16);                // tensor_dim0[15:0]
  g1[2] = (int)(((tensor_d0 >> 16) & 0xFFFFu) |
                ((tensor_d1 & 0xFFFFu) << 16));              // dim0[31:16] | dim1[15:0]
  g1[3] = (int)(((tensor_d1 >> 16) & 0xFFFFu) | (32u << 16));// dim1[31:16] | tile_dim0=32
  g1[4] = (int)(tile_rows & 0xFFFFu);                        // tile_dim1 (tile_dim2=0)
  g1[5] = (int)ldk;                                          // tensor_dim0_stride[31:0]
  g1[6] = 0;                                                 // stride[47:32]=0, dim1_stride lo=0
  g1[7] = 0;
  const i32x4 z4 = {0, 0, 0, 0};
#if __clang_major__ >= 23
  const i32x8 z8 = {0, 0, 0, 0, 0, 0, 0, 0};
  __builtin_amdgcn_tensor_load_to_lds(g0, g1, z4, z4, z8, 0);
#else
  __builtin_amdgcn_tensor_load_to_lds(g0, g1, z4, z4, 0);
#endif
#endif
}

// =====================================================================
// LDS-tiled, TDM double-buffered GEMM: C[M,N] = A[M,K] * Bt[N,K]^T
// Block: 256 thr / 8 waves; block tile 64x128; wave tile 32x32 (4 WMMA).
// Requires M%64==0, N%128==0, K%32==0.
// =====================================================================
#define MBLK   64
#define NBLK   128
#define KSTEP  32
#define A_TB   (MBLK * KSTEP * 2)            // 4 KB
#define B_TB   (NBLK * KSTEP * 2)            // 8 KB
#define BUF_TB (A_TB + B_TB)                 // 12 KB (x2 buffers = 24 KB LDS)

__global__ void gemm_bf16_wmma_tdm(const bf16_t* __restrict__ A, const bf16_t* __restrict__ Bt,
                                   float* __restrict__ C, int M, int N, int K) {
  __shared__ __align__(16) char smem[2 * BUF_TB];
  // Generic pointer to LDS: low 32 bits ARE the LDS byte offset (ISA 10.2 aperture map).
  const unsigned int lds_base = (unsigned int)(uintptr_t)(void*)smem;

  const int wid  = threadIdx.x >> 5;
  const int lane = threadIdx.x & 31;
  const int nBlocks = N / NBLK;
  const int mB = blockIdx.x / nBlocks;
  const int nB = blockIdx.x - mB * nBlocks;
  const int row0 = mB * MBLK;   // A rows of this block tile
  const int col0 = nB * NBLK;   // Bt rows (= C cols) of this block tile
  const int mw = wid & 1;       // 0..1 -> 32-row slice
  const int nw = wid >> 1;      // 0..3 -> 32-col slice
  const int nK = K / KSTEP;

  v8f acc[2][2];
#pragma unroll
  for (int i = 0; i < 2; ++i)
#pragma unroll
    for (int j = 0; j < 2; ++j) acc[i][j] = v8f{0.f, 0.f, 0.f, 0.f, 0.f, 0.f, 0.f, 0.f};

#if HAVE_TDM
  if (wid == 0) {  // preload tile 0 into buffer 0 (wave-level DMA, EXEC-agnostic)
    tdm_load_tile(lds_base,        A  + (size_t)row0 * K, K, M, MBLK, K);
    tdm_load_tile(lds_base + A_TB, Bt + (size_t)col0 * K, K, N, NBLK, K);
  }
#endif

  for (int i = 0; i < nK; ++i) {
    const int cur = i & 1;
#if HAVE_TDM
    if (wid == 0) {
      if (i + 1 < nK) {  // kick DMA for next K-slab into the other buffer
        const int k1 = (i + 1) * KSTEP;
        const unsigned int b = lds_base + (unsigned int)((cur ^ 1) * BUF_TB);
        tdm_load_tile(b,        A  + (size_t)row0 * K + k1, K, M, MBLK, K);
        tdm_load_tile(b + A_TB, Bt + (size_t)col0 * K + k1, K, N, NBLK, K);
        __builtin_amdgcn_s_wait_tensorcnt(2);  // tile i's pair has landed
      } else {
        __builtin_amdgcn_s_wait_tensorcnt(0);
      }
    }
    __syncthreads();  // LDS tile i visible to all 8 waves
#else
    {  // fallback: synchronous cooperative staging of tile i
      const int k0 = i * KSTEP;
      char* bufAw = smem + cur * BUF_TB;
      char* bufBw = bufAw + A_TB;
      {
        const int r = threadIdx.x >> 2, cb = (threadIdx.x & 3) * 16;
        *(uint4*)(bufAw + threadIdx.x * 16) =
            *(const uint4*)((const char*)A + ((size_t)(row0 + r) * K + k0) * 2 + cb);
      }
#pragma unroll
      for (int j = 0; j < 2; ++j) {
        const int idx = threadIdx.x + j * 256;
        const int r = idx >> 2, cb = (idx & 3) * 16;
        *(uint4*)(bufBw + idx * 16) =
            *(const uint4*)((const char*)Bt + ((size_t)(col0 + r) * K + k0) * 2 + cb);
      }
    }
    __syncthreads();
#endif
    const char* bufA = smem + cur * BUF_TB;
    const char* bufB = bufA + A_TB;
    const v16bf a0 = frag_a_from(bufA + (size_t)(mw * 32 +  0) * (KSTEP * 2), KSTEP * 2);
    const v16bf a1 = frag_a_from(bufA + (size_t)(mw * 32 + 16) * (KSTEP * 2), KSTEP * 2);
    const v16bf b0 = frag_b_from(bufB + (size_t)(nw * 32 +  0) * (KSTEP * 2), KSTEP * 2);
    const v16bf b1 = frag_b_from(bufB + (size_t)(nw * 32 + 16) * (KSTEP * 2), KSTEP * 2);
    acc[0][0] = __builtin_amdgcn_wmma_f32_16x16x32_bf16(false, a0, false, b0, (short)0, acc[0][0], false, false);
    acc[0][1] = __builtin_amdgcn_wmma_f32_16x16x32_bf16(false, a0, false, b1, (short)0, acc[0][1], false, false);
    acc[1][0] = __builtin_amdgcn_wmma_f32_16x16x32_bf16(false, a1, false, b0, (short)0, acc[1][0], false, false);
    acc[1][1] = __builtin_amdgcn_wmma_f32_16x16x32_bf16(false, a1, false, b1, (short)0, acc[1][1], false, false);
    __syncthreads();  // all waves done reading buf[cur] before it is refilled
  }

  // C/D f32 layout: VGPR r -> row r + 8*half; lane&15 -> col
#pragma unroll
  for (int ms = 0; ms < 2; ++ms) {
    const int mBase = row0 + mw * 32 + ms * 16 + ((lane >> 4) << 3);
#pragma unroll
    for (int ns = 0; ns < 2; ++ns) {
      const int n = col0 + nw * 32 + ns * 16 + (lane & 15);
#pragma unroll
      for (int r = 0; r < 8; ++r)
        C[(size_t)(mBase + r) * N + n] = acc[ms][ns][r];
    }
  }
}

// =====================================================================
// Simple per-wave-tile GEMM (kept for N=96 GEMM where 128-wide tiles
// don't divide): C[M,N] = A[M,K] * Bt[N,K]^T, one 16x16 tile per wave.
// =====================================================================
__global__ void gemm_bf16_wmma(const bf16_t* __restrict__ A, const bf16_t* __restrict__ Bt,
                               float* __restrict__ C, int M, int N, int K) {
  const int wid    = threadIdx.x >> 5;
  const int tile   = blockIdx.x * (blockDim.x >> 5) + wid;
  const int nTiles = N >> 4;
  const int mT = tile / nTiles;
  const int nT = tile - mT * nTiles;
  if (mT >= (M >> 4)) return;  // wave-uniform guard

  v8f acc = {0.f, 0.f, 0.f, 0.f, 0.f, 0.f, 0.f, 0.f};
  for (int k0 = 0; k0 < K; k0 += 32) {
    if (k0 + 32 < K) {
      __builtin_prefetch(A + (size_t)((mT << 4) + (threadIdx.x & 15)) * K + k0 + 32, 0, 1);
    }
    const v16bf a = frag_a_from((const char*)A + ((size_t)(mT << 4) * K + k0) * 2, K * 2);
    const v16bf b = frag_b_from((const char*)Bt + ((size_t)(nT << 4) * K + k0) * 2, K * 2);
    acc = __builtin_amdgcn_wmma_f32_16x16x32_bf16(false, a, false, b, (short)0, acc, false, false);
  }
  const int lane  = threadIdx.x & 31;
  const int n     = (nT << 4) + (lane & 15);
  const int mBase = (mT << 4) + ((lane >> 4) << 3);
#pragma unroll
  for (int r = 0; r < 8; ++r)
    C[(size_t)(mBase + r) * N + n] = acc[r];
}

// ---------------- block reduction helper (wave32-aware) ----------------
__device__ __forceinline__ float block_reduce(float v, float* sred) {
#pragma unroll
  for (int off = 16; off > 0; off >>= 1) v += __shfl_down(v, off, 32);
  const int lane = threadIdx.x & 31, w = threadIdx.x >> 5;
  if (lane == 0) sred[w] = v;
  __syncthreads();
  float t = (threadIdx.x < 8) ? sred[threadIdx.x] : 0.0f;
  if (w == 0) {
#pragma unroll
    for (int off = 4; off > 0; off >>= 1) t += __shfl_down(t, off, 32);
    if (lane == 0) sred[8] = t;
  }
  __syncthreads();
  return sred[8];
}

// ---------------- fused residual-add + LayerNorm ----------------
__global__ void add_ln_kernel(const float* __restrict__ x, const float* __restrict__ res,
                              const float* __restrict__ gamma, const float* __restrict__ beta,
                              float* __restrict__ resid_out, bf16_t* __restrict__ xn) {
  __shared__ float sred[9];
  const int row = blockIdx.x;
  const float* xr = x   + (size_t)row * DM;
  const float* rr = res + (size_t)row * DM;
  float*  ro = resid_out + (size_t)row * DM;
  bf16_t* xo = xn        + (size_t)row * DM;
  float v[4];
  float s = 0.f;
#pragma unroll
  for (int i = 0; i < 4; ++i) {
    const int c = threadIdx.x + (i << 8);
    v[i] = xr[c] + rr[c];
    ro[c] = v[i];
    s += v[i];
  }
  const float mu = block_reduce(s, sred) * (1.0f / DM);
  float vs = 0.f;
#pragma unroll
  for (int i = 0; i < 4; ++i) { const float d = v[i] - mu; vs += d * d; }
  const float rstd = rsqrtf(block_reduce(vs, sred) * (1.0f / DM) + 1e-5f);
#pragma unroll
  for (int i = 0; i < 4; ++i) {
    const int c = threadIdx.x + (i << 8);
    xo[c] = (bf16_t)((v[i] - mu) * rstd * gamma[c] + beta[c]);
  }
}

// ---------------- transpose + f32->bf16 weight conversion ----------------
__global__ void wt_convert_kernel(const float* __restrict__ W, bf16_t* __restrict__ Wt,
                                  int K, int N) {
  const int idx = blockIdx.x * blockDim.x + threadIdx.x;
  if (idx >= K * N) return;
  const int k = idx / N, n = idx - k * N;
  Wt[(size_t)n * K + k] = (bf16_t)W[idx];
}

// ---------------- causal depthwise conv (width 4) + SiLU ----------------
__global__ void conv_silu_kernel(const float* __restrict__ xz, const float* __restrict__ cw,
                                 const float* __restrict__ cb, float* __restrict__ xc,
                                 bf16_t* __restrict__ xcb) {
  const int idx = blockIdx.x * blockDim.x + threadIdx.x;  // Rn*DI
  const int row = idx >> 11;
  const int d   = idx & (DI - 1);
  const int t   = row & (Lx - 1);
  float a = cb[d];
#pragma unroll
  for (int k = 0; k < DCV; ++k) {
    const int tt = t - (DCV - 1) + k;
    if (tt >= 0)
      a += cw[d * DCV + k] * xz[(size_t)(row - (DCV - 1) + k) * (2 * DI) + d];
  }
  a = a / (1.f + __expf(-a));  // SiLU
  xc[idx]  = a;
  xcb[idx] = (bf16_t)a;
}

// ---------------- slice dt columns of xdbl, convert to bf16 ----------------
__global__ void slice_dt_kernel(const float* __restrict__ xdbl, bf16_t* __restrict__ dtb) {
  const int idx = blockIdx.x * blockDim.x + threadIdx.x;  // Rn*DTR
  const int row = idx >> 6;
  const int k   = idx & (DTR - 1);
  dtb[idx] = (bf16_t)xdbl[(size_t)row * NXD + k];
}

// ---------------- delta = softplus(gemm + b_dt), in place ----------------
__global__ void softplus_bias_kernel(float* __restrict__ delta, const float* __restrict__ b) {
  const int idx = blockIdx.x * blockDim.x + threadIdx.x;  // Rn*DI
  const int d   = idx & (DI - 1);
  const float v = delta[idx] + b[d];
  delta[idx] = (v > 20.f) ? v : log1pf(__expf(v));
}

// ---------------- selective scan: one (b,d) channel per lane ----------------
__global__ void ssm_scan_kernel(const float* __restrict__ delta, const float* __restrict__ xc,
                                const float* __restrict__ xdbl, const float* __restrict__ A_log,
                                float* __restrict__ yraw) {
  const int idx = blockIdx.x * blockDim.x + threadIdx.x;  // Bx*DI = 4096
  const int b = idx >> 11;
  const int d = idx & (DI - 1);
  float Ac[DST], h[DST];
#pragma unroll
  for (int n = 0; n < DST; ++n) {
    Ac[n] = -__expf(A_log[d * DST + n]);
    h[n]  = 0.f;
  }
  for (int t = 0; t < Lx; ++t) {
    const size_t row = (size_t)b * Lx + t;
    const float dl = delta[row * DI + d];
    const float xt = xc[row * DI + d];
    const float dx = dl * xt;
    const float* bc = xdbl + row * NXD;  // B at [64..79], C at [80..95]
    float y = 0.f;
#pragma unroll
    for (int n = 0; n < DST; ++n) {
      h[n] = __expf(dl * Ac[n]) * h[n] + dx * bc[DTR + n];
      y = fmaf(h[n], bc[DTR + DST + n], y);
    }
    yraw[row * DI + d] = y;
  }
}

// ---------------- y = (yraw + xc*D) * silu(z) -> bf16 ----------------
__global__ void gate_kernel(const float* __restrict__ yraw, const float* __restrict__ xc,
                            const float* __restrict__ xz, const float* __restrict__ Dskip,
                            bf16_t* __restrict__ yb) {
  const int idx = blockIdx.x * blockDim.x + threadIdx.x;  // Rn*DI
  const int row = idx >> 11;
  const int d   = idx & (DI - 1);
  const float z = xz[(size_t)row * (2 * DI) + DI + d];
  float y = yraw[idx] + xc[idx] * Dskip[d];
  y *= z / (1.f + __expf(-z));
  yb[idx] = (bf16_t)y;
}

// ---------------- host-side launch ----------------
extern "C" void kernel_launch(void* const* d_in, const int* in_sizes, int n_in,
                              void* d_out, int out_size, void* d_ws, size_t ws_size,
                              hipStream_t stream) {
  const float* x      = (const float*)d_in[0];
  const float* resid  = (const float*)d_in[1];
  const float* gamma  = (const float*)d_in[2];
  const float* beta   = (const float*)d_in[3];
  const float* W_in   = (const float*)d_in[4];
  const float* conv_w = (const float*)d_in[5];
  const float* conv_b = (const float*)d_in[6];
  const float* W_x    = (const float*)d_in[7];
  const float* W_dt   = (const float*)d_in[8];
  const float* b_dt   = (const float*)d_in[9];
  const float* A_log  = (const float*)d_in[10];
  const float* D_skip = (const float*)d_in[11];
  const float* W_out  = (const float*)d_in[12];

  float* out_hidden = (float*)d_out;                       // (B,L,DM)
  float* out_resid  = out_hidden + (size_t)Rn * DM;        // (B,L,DM)

  char* ws = (char*)d_ws;
  bf16_t* xn      = (bf16_t*)(ws + OFF_XN);
  bf16_t* WtIn    = (bf16_t*)(ws + OFF_WTIN);
  bf16_t* WtX     = (bf16_t*)(ws + OFF_WTX);
  bf16_t* WtDt    = (bf16_t*)(ws + OFF_WTDT);
  bf16_t* WtOut   = (bf16_t*)(ws + OFF_WTO);
  float*  xz      = (float*) (ws + OFF_XZ);
  float*  xc      = (float*) (ws + OFF_XC);
  bf16_t* xcb     = (bf16_t*)(ws + OFF_XCB);
  float*  xdbl    = (float*) (ws + OFF_XDBL);
  bf16_t* dtb     = (bf16_t*)(ws + OFF_DTB);
  float*  deltaB  = (float*) (ws + OFF_DELTA);
  float*  yrawB   = (float*) (ws + OFF_YRAW);
  bf16_t* yb      = (bf16_t*)(ws + OFF_YB);

  const int TB = 256;

  // 1) residual add + LayerNorm (also writes the resid output)
  add_ln_kernel<<<Rn, TB, 0, stream>>>(x, resid, gamma, beta, out_resid, xn);

  // 2) bf16-transpose all weights
  wt_convert_kernel<<<(DM * 2 * DI + TB - 1) / TB, TB, 0, stream>>>(W_in,  WtIn,  DM, 2 * DI);
  wt_convert_kernel<<<(DI * NXD   + TB - 1) / TB, TB, 0, stream>>>(W_x,   WtX,   DI, NXD);
  wt_convert_kernel<<<(DTR * DI   + TB - 1) / TB, TB, 0, stream>>>(W_dt,  WtDt,  DTR, DI);
  wt_convert_kernel<<<(DI * DM    + TB - 1) / TB, TB, 0, stream>>>(W_out, WtOut, DI, DM);

  // 3) xz = xn @ W_in   (4096 x 1024 x 4096) -- TDM + LDS tiled
  gemm_bf16_wmma_tdm<<<(Rn / MBLK) * ((2 * DI) / NBLK), TB, 0, stream>>>(
      xn, WtIn, xz, Rn, 2 * DI, DM);

  // 4) depthwise causal conv + SiLU -> xc (f32 + bf16)
  conv_silu_kernel<<<(Rn * DI + TB - 1) / TB, TB, 0, stream>>>(xz, conv_w, conv_b, xc, xcb);

  // 5) xdbl = xc @ W_x  (4096 x 2048 x 96) -- narrow N, simple kernel
  {
    const int tiles = (Rn >> 4) * (NXD >> 4);
    gemm_bf16_wmma<<<(tiles + 7) / 8, TB, 0, stream>>>(xcb, WtX, xdbl, Rn, NXD, DI);
  }

  // 6) dt slice -> bf16, then delta_raw = dt @ W_dt (4096 x 64 x 2048)
  slice_dt_kernel<<<(Rn * DTR + TB - 1) / TB, TB, 0, stream>>>(xdbl, dtb);
  gemm_bf16_wmma_tdm<<<(Rn / MBLK) * (DI / NBLK), TB, 0, stream>>>(
      dtb, WtDt, deltaB, Rn, DI, DTR);

  // 7) delta = softplus(delta_raw + b_dt)
  softplus_bias_kernel<<<(Rn * DI + TB - 1) / TB, TB, 0, stream>>>(deltaB, b_dt);

  // 8) selective scan: 4096 channels in parallel, sequential over L
  ssm_scan_kernel<<<(Bx * DI) / TB, TB, 0, stream>>>(deltaB, xc, xdbl, A_log, yrawB);

  // 9) gating: y = (yraw + xc*D) * silu(z) -> bf16
  gate_kernel<<<(Rn * DI + TB - 1) / TB, TB, 0, stream>>>(yrawB, xc, xz, D_skip, yb);

  // 10) hidden = y @ W_out (4096 x 2048 x 1024), straight into d_out
  gemm_bf16_wmma_tdm<<<(Rn / MBLK) * (DM / NBLK), TB, 0, stream>>>(
      yb, WtOut, out_hidden, Rn, DM, DI);
}
